// SelfModifyingLayer_23210003268111
// MI455X (gfx1250) — matrix operand order repaired and checked by reading, not compiled, not run on hardware
//
#include <hip/hip_runtime.h>
#include <stdint.h>

// Problem constants (fixed by the reference)
#define B_DIM   8
#define S_DIM   4096
#define D_DIM   1024
#define M_TOTAL (B_DIM * S_DIM)      // 32768
#define CHUNK   128
#define NCHUNK  (S_DIM / CHUNK)      // 32

// GEMM tiling
#define BLK_M 128
#define BLK_N 128
#define BLK_K 32
#define LDSTR 40                     // padded LDS row stride in bf16 elems (80 B, 16B-aligned)
#define KSTEPS (D_DIM / BLK_K)       // 32

typedef __attribute__((ext_vector_type(16))) __bf16 v16bf;
typedef __attribute__((ext_vector_type(8)))  float  v8f;

union Frag16 { v16bf v; uint4 q[2]; };

#if defined(__gfx1250__) && __has_builtin(__builtin_amdgcn_global_load_async_to_lds_b128)
#define HAVE_ASYNC_LDS 1
typedef int v4i __attribute__((vector_size(16)));
typedef __attribute__((address_space(1))) v4i as1_v4i;
typedef __attribute__((address_space(3))) v4i as3_v4i;
#endif

__device__ __forceinline__ void wait_async_zero() {
#if defined(HAVE_ASYNC_LDS)
#if __has_builtin(__builtin_amdgcn_s_wait_asynccnt)
    __builtin_amdgcn_s_wait_asynccnt(0);
#else
    asm volatile("s_wait_asynccnt 0x0" ::: "memory");
#endif
#endif
}

__device__ __forceinline__ unsigned short f2bf_bits(float f) {
    union { float f; uint32_t u; } cv; cv.f = f;
    uint32_t u = cv.u + 0x7FFFu + ((cv.u >> 16) & 1u);   // round-to-nearest-even
    return (unsigned short)(u >> 16);
}

__device__ __forceinline__ uint32_t pack_bf16(float x, float y) {
#if defined(__gfx1250__) && __has_builtin(__builtin_amdgcn_cvt_pk_bf16_f32)
    typedef __attribute__((ext_vector_type(2))) __bf16 v2bf;
    union { v2bf v; uint32_t u; } r;
    r.v = __builtin_amdgcn_cvt_pk_bf16_f32(x, y);
    return r.u;
#else
    return (uint32_t)f2bf_bits(x) | ((uint32_t)f2bf_bits(y) << 16);
#endif
}

// ---------------------------------------------------------------------------
// Weight conversion: fp32 -> bf16 bits (once, 1M elems each)
// ---------------------------------------------------------------------------
__global__ __launch_bounds__(256) void convert_weights(
    const float* __restrict__ Wu, const float* __restrict__ Wf,
    unsigned short* __restrict__ Wub, unsigned short* __restrict__ Wfb)
{
    int i = blockIdx.x * 256 + threadIdx.x;
    if (i < D_DIM * D_DIM) {
        Wub[i] = f2bf_bits(Wu[i]);
        Wfb[i] = f2bf_bits(Wf[i]);
    }
}

// ---------------------------------------------------------------------------
// C[M,N] = A[M,K] (fp32 -> bf16 in-register) x W[N,K]^T (bf16) + bias
// 256 threads = 8 waves arranged 4(M) x 2(N); each wave owns 32 rows x 64 cols
// of the 128x128 block tile: 2 A-frags x 4 B-frags = 8 WMMAs per K-step.
// Double-buffered LDS; B tile staged via global_load_async_to_lds_b128.
// ---------------------------------------------------------------------------
__global__ __launch_bounds__(256) void gemm_bf16_wmma(
    const float* __restrict__ A,
    const unsigned short* __restrict__ Wb,
    const float* __restrict__ bias,
    float* __restrict__ Out,
    int M, int N, int K)
{
    __shared__ unsigned short As[2][BLK_M * LDSTR];   // [m][k], 10 KB each
    __shared__ unsigned short Bs[2][BLK_N * LDSTR];   // [n][k] == row-major W slice

    const int tid  = threadIdx.x;
    const int lane = tid & 31;
    const int wave = tid >> 5;          // 0..7
    const int wm   = wave >> 1;         // 0..3  (M group: 32 rows)
    const int wn   = wave & 1;          // 0..1  (N group: 64 cols)
    const int hlf  = lane >> 4;         // 0/1 (lane half)
    const int mlan = lane & 15;         // 0..15

    const int m_base = blockIdx.y * BLK_M;
    const int n_base = blockIdx.x * BLK_N;

    // cooperative tile-load indices: thread t -> row t/2, k-half (t&1)*16
    const int lr = tid >> 1;
    const int lk = (tid & 1) * 16;

    v8f acc[2][4] = {};

    // ---- tile stager: global -> LDS buffer `buf` for K offset `kk` ----
    auto stage_tile = [&](int buf, int kk) {
        // A: fp32 -> bf16 in-register, ds_store_b128 x2
        const float* ag = A + (size_t)(m_base + lr) * K + kk + lk;
        float4 f0 = *(const float4*)(ag + 0);
        float4 f1 = *(const float4*)(ag + 4);
        float4 f2 = *(const float4*)(ag + 8);
        float4 f3 = *(const float4*)(ag + 12);
        uint32_t p0 = pack_bf16(f0.x, f0.y);
        uint32_t p1 = pack_bf16(f0.z, f0.w);
        uint32_t p2 = pack_bf16(f1.x, f1.y);
        uint32_t p3 = pack_bf16(f1.z, f1.w);
        uint32_t p4 = pack_bf16(f2.x, f2.y);
        uint32_t p5 = pack_bf16(f2.z, f2.w);
        uint32_t p6 = pack_bf16(f3.x, f3.y);
        uint32_t p7 = pack_bf16(f3.z, f3.w);
        uint4* aw = (uint4*)&As[buf][lr * LDSTR + lk];
        aw[0] = make_uint4(p0, p1, p2, p3);
        aw[1] = make_uint4(p4, p5, p6, p7);
        if (kk + BLK_K < K) __builtin_prefetch(ag + BLK_K, 0, 1);   // warm L2 for A

        // B: bf16 weights, straight global -> LDS (ASYNCcnt path on gfx1250)
        const unsigned short* bgp = Wb + (size_t)(n_base + lr) * K + kk + lk;
        unsigned short* bdst = &Bs[buf][lr * LDSTR + lk];
#if defined(HAVE_ASYNC_LDS)
        __builtin_amdgcn_global_load_async_to_lds_b128(
            (as1_v4i*)(void*)const_cast<unsigned short*>(bgp),
            (as3_v4i*)(void*)bdst, 0, 0);
        __builtin_amdgcn_global_load_async_to_lds_b128(
            (as1_v4i*)(void*)const_cast<unsigned short*>(bgp + 8),
            (as3_v4i*)(void*)(bdst + 8), 0, 0);
#else
        const uint4* bg = (const uint4*)bgp;
        uint4* bw = (uint4*)bdst;
        bw[0] = bg[0];
        bw[1] = bg[1];
#endif
    };

    // prologue: stage tile 0
    stage_tile(0, 0);

    for (int step = 0; step < KSTEPS; ++step) {
        const int cur = step & 1;

        wait_async_zero();      // own async copies of tile `step` landed in LDS
        __syncthreads();        // tile `step` visible block-wide; prev compute done

        if (step + 1 < KSTEPS) stage_tile(cur ^ 1, (step + 1) * BLK_K);

        // ---- 2 A fragments (contiguous K: one 16-wide K run per lane, matching B) ----
        Frag16 af[2];
        #pragma unroll
        for (int mi = 0; mi < 2; ++mi) {
            const uint4* arow =
                (const uint4*)&As[cur][(wm * 32 + mi * 16 + mlan) * LDSTR];
            af[mi].q[0] = arow[2 * hlf];        // K = hlf*16 .. +8
            af[mi].q[1] = arow[2 * hlf + 1];    // K = hlf*16+8 .. +8
        }
        // ---- 4 B fragments ----
        Frag16 bfr[4];
        #pragma unroll
        for (int nj = 0; nj < 4; ++nj) {
            const uint4* brow =
                (const uint4*)&Bs[cur][(wn * 64 + nj * 16 + mlan) * LDSTR + hlf * 16];
            bfr[nj].q[0] = brow[0];
            bfr[nj].q[1] = brow[1];
        }
        // ---- 8 WMMAs: A frags reused 4x, B frags reused 2x ----
        #pragma unroll
        for (int mi = 0; mi < 2; ++mi) {
            #pragma unroll
            for (int nj = 0; nj < 4; ++nj) {
                acc[mi][nj] = __builtin_amdgcn_wmma_f32_16x16x32_bf16(
                    false, af[mi].v, false, bfr[nj].v, (short)0, acc[mi][nj],
                    false, false);
            }
        }
    }

    // ---- epilogue: bias + fp32 store (C layout: VGPR r -> M=hlf*8+r, N=mlan) ----
    #pragma unroll
    for (int mi = 0; mi < 2; ++mi) {
        #pragma unroll
        for (int nj = 0; nj < 4; ++nj) {
            const int ncol = n_base + wn * 64 + nj * 16 + mlan;
            const float bv = bias[ncol];
            #pragma unroll
            for (int r = 0; r < 8; ++r) {
                const int row = m_base + wm * 32 + mi * 16 + hlf * 8 + r;
                Out[(size_t)row * N + ncol] = acc[mi][nj][r] + bv;
            }
        }
    }
}

// ---------------------------------------------------------------------------
// Phase 1: in-place local affine scans over 128-step chunks.
// thread = (b, chunk, d); lane d==0 of each (b,chunk) records prefix products.
// ---------------------------------------------------------------------------
__global__ __launch_bounds__(256) void scan_local(
    float* __restrict__ buf,             // [B,S,D] upd -> local mem (in place)
    const float* __restrict__ mask,      // [B,S]
    const float* __restrict__ decay_param,
    float* __restrict__ pp)              // [B,S] inclusive within-chunk prefix prod
{
    const int tid = blockIdx.x * 256 + threadIdx.x;       // B*NCHUNK*D
    const int d = tid % D_DIM;
    const int c = (tid / D_DIM) % NCHUNK;
    const int b = tid / (D_DIM * NCHUNK);
    const float decay = 1.0f / (1.0f + __expf(-decay_param[0]));

    const int s0 = c * CHUNK;
    float mem = 0.0f, prod = 1.0f;
    for (int s = 0; s < CHUNK; ++s) {
        const float m  = mask[b * S_DIM + s0 + s];
        const float a  = m * decay + (1.0f - m);
        const size_t i = ((size_t)(b * S_DIM + s0 + s)) * D_DIM + d;
        const float bb = m * buf[i];
        mem = fmaf(a, mem, bb);
        buf[i] = mem;
        prod *= a;
        if (d == 0) pp[b * S_DIM + s0 + s] = prod;
    }
}

// ---------------------------------------------------------------------------
// Phase 2: propagate carries across chunks (sequential over 32 chunks).
// ---------------------------------------------------------------------------
__global__ __launch_bounds__(256) void scan_carry(
    const float* __restrict__ buf, const float* __restrict__ pp,
    float* __restrict__ carry)           // [B, NCHUNK, D] carry INTO each chunk
{
    const int tid = blockIdx.x * 256 + threadIdx.x;       // B*D
    const int d = tid % D_DIM;
    const int b = tid / D_DIM;
    float c = 0.0f;
    for (int ch = 0; ch < NCHUNK; ++ch) {
        const int send = ch * CHUNK + CHUNK - 1;
        const float P = pp[b * S_DIM + send];
        const float e = buf[((size_t)(b * S_DIM + send)) * D_DIM + d];
        carry[((size_t)(b * NCHUNK + ch)) * D_DIM + d] = c;
        c = fmaf(P, c, e);
    }
}

// ---------------------------------------------------------------------------
// Phase 3: mem[s] = local[s] + pp[s] * carry_in[chunk(s)]   (in place)
// ---------------------------------------------------------------------------
__global__ __launch_bounds__(256) void scan_fixup(
    float* __restrict__ buf, const float* __restrict__ pp,
    const float* __restrict__ carry)
{
    const size_t idx = (size_t)blockIdx.x * 256 + threadIdx.x;  // B*S*D
    const int d = (int)(idx % D_DIM);
    const int s = (int)((idx / D_DIM) % S_DIM);
    const int b = (int)(idx / ((size_t)D_DIM * S_DIM));
    const int ch = s / CHUNK;
    buf[idx] = fmaf(pp[b * S_DIM + s],
                    carry[((size_t)(b * NCHUNK + ch)) * D_DIM + d],
                    buf[idx]);
}

// ---------------------------------------------------------------------------
extern "C" void kernel_launch(void* const* d_in, const int* in_sizes, int n_in,
                              void* d_out, int out_size, void* d_ws, size_t ws_size,
                              hipStream_t stream) {
    const float* x     = (const float*)d_in[0];   // [B,S,D]
    const float* mask  = (const float*)d_in[1];   // [B,S]
    const float* Wu    = (const float*)d_in[2];   // [D,D]
    const float* bu    = (const float*)d_in[3];   // [D]
    const float* Wf    = (const float*)d_in[4];   // [D,D]
    const float* bfv   = (const float*)d_in[5];   // [D]
    const float* decay = (const float*)d_in[6];   // [1]
    float* out = (float*)d_out;

    // workspace layout
    float*          buf  = (float*)d_ws;                                   // 128 MB
    unsigned short* Wub  = (unsigned short*)(buf + (size_t)M_TOTAL * D_DIM);
    unsigned short* Wfb  = Wub + (size_t)D_DIM * D_DIM;
    float*          pp   = (float*)(Wfb + (size_t)D_DIM * D_DIM);          // B*S
    float*          cry  = pp + (size_t)B_DIM * S_DIM;                     // B*NCHUNK*D

    // 1) weights -> bf16
    convert_weights<<<(D_DIM * D_DIM + 255) / 256, 256, 0, stream>>>(Wu, Wf, Wub, Wfb);

    // 2) upd = x @ Wu^T + bu
    dim3 ggrid(D_DIM / BLK_N, M_TOTAL / BLK_M);
    gemm_bf16_wmma<<<ggrid, 256, 0, stream>>>(x, Wub, bu, buf, M_TOTAL, D_DIM, D_DIM);

    // 3) chunked affine scan (in place on buf)
    scan_local<<<(B_DIM * NCHUNK * D_DIM) / 256, 256, 0, stream>>>(buf, mask, decay, pp);
    scan_carry<<<(B_DIM * D_DIM) / 256, 256, 0, stream>>>(buf, pp, cry);
    scan_fixup<<<(unsigned)(((size_t)M_TOTAL * D_DIM) / 256), 256, 0, stream>>>(buf, pp, cry);

    // 4) out = mem @ Wf^T + bf
    gemm_bf16_wmma<<<ggrid, 256, 0, stream>>>(buf, Wfb, bfv, out, M_TOTAL, D_DIM, D_DIM);
}
